// RecurrentEntitiyDecoder_54382875902324
// MI455X (gfx1250) — matrix operand order repaired and checked by reading, not compiled, not run on hardware
//
#include <hip/hip_runtime.h>
#include <hip/hip_bf16.h>
#include <stdint.h>

// Problem constants (match reference)
#define VOCAB 50257
#define DIM   1024
#define FC    2048
#define BATCH 64
#define ENT   512

typedef __attribute__((ext_vector_type(2)))  float  v2f;
typedef __attribute__((ext_vector_type(8)))  float  v8f;
typedef __attribute__((ext_vector_type(16))) __bf16 v16bf;

// ---------------------------------------------------------------------------
// Stage 1a: masked entity reduction, partial sums (deterministic, no atomics)
// grid (DIM/256, 64), block 256. part[y][d] = sum over rows [y*512,(y+1)*512)
// entity_hiddens is a 128MB single-use stream -> non-temporal loads.
// The mask test is block-uniform (r independent of tid) -> scalar branch,
// skipping the load entirely for masked-out rows.
// ---------------------------------------------------------------------------
__global__ void k_ent_partial(const float* __restrict__ eh,
                              const uint8_t* __restrict__ km,
                              float* __restrict__ part) {
    const int d  = blockIdx.x * 256 + threadIdx.x;
    const int r0 = blockIdx.y * 512;
    float acc = 0.f;
    for (int r = r0; r < r0 + 512; ++r) {
        if (km[r]) acc += __builtin_nontemporal_load(&eh[(size_t)r * DIM + d]);
    }
    part[(size_t)blockIdx.y * DIM + d] = acc;
}

// Stage 1b: fold 64 partials in fixed order -> ent_sum[D]
__global__ void k_ent_reduce(const float* __restrict__ part,
                             float* __restrict__ ent_sum) {
    const int d = blockIdx.x * 256 + threadIdx.x;
    float s = 0.f;
    for (int y = 0; y < 64; ++y) s += part[(size_t)y * DIM + d];
    ent_sum[d] = s;
}

// ---------------------------------------------------------------------------
// Small GEMMs (M=64) with V_WMMA_F32_16X16X4_F32, one 16x16 tile per wave.
// MODE 0: A = emb[question[row]]   epilogue: relu(acc + bias[n])
// MODE 1: A = Amat                 epilogue: acc + bias[n]
// MODE 2: A = Amat                 epilogue: acc * extra[n]         (ent_sum)
// MODE 3: A = Amat                 epilogue: acc + extra[row*N + n] (add q)
// Fragment layouts per CDNA5 ISA §7.12.2 (32-bit A 16x4 / B 4x16 / C 16x16).
// ---------------------------------------------------------------------------
template <int MODE>
__global__ void k_gemm64(const float* __restrict__ Amat,
                         const float* __restrict__ Bmat,
                         const float* __restrict__ bias,
                         const float* __restrict__ extra,
                         const int*   __restrict__ question,
                         const float* __restrict__ emb,
                         float* __restrict__ out, int K, int N) {
    const int lane   = threadIdx.x & 31;
    const int wid    = blockIdx.x * (blockDim.x >> 5) + (threadIdx.x >> 5);
    const int ntiles = N >> 4;
    const int mtile  = wid & 3;
    const int ntile  = wid >> 2;
    if (ntile >= ntiles) return;           // wave-uniform

    const int half = lane >> 4;            // 0: lanes 0-15, 1: lanes 16-31
    const int lm   = lane & 15;
    const int n    = ntile * 16 + lm;      // output column for this lane
    const int row  = mtile * 16 + lm;      // A row for this lane

    const float* arow;
    if (MODE == 0) arow = emb + (size_t)question[row] * DIM;
    else           arow = Amat + (size_t)row * K;

    v8f acc0 = {}; v8f acc1 = {};          // two chains for WMMA ILP
    for (int k = 0; k < K; k += 8) {
        v2f a, b;
        // A 16x4: VGPR0 = K(2*half), VGPR1 = K(2*half+1)
        a[0] = arow[k + 2 * half];
        a[1] = arow[k + 2 * half + 1];
        // B 4x16: VGPR0 = row K(2*half), VGPR1 = row K(2*half+1), lane = N
        b[0] = Bmat[(size_t)(k + 2 * half)     * N + n];
        b[1] = Bmat[(size_t)(k + 2 * half + 1) * N + n];
        acc0 = __builtin_amdgcn_wmma_f32_16x16x4_f32(false, a, false, b,
                                                     (short)0, acc0, false, false);
        const int k2 = k + 4;
        v2f a2, b2;
        a2[0] = arow[k2 + 2 * half];
        a2[1] = arow[k2 + 2 * half + 1];
        b2[0] = Bmat[(size_t)(k2 + 2 * half)     * N + n];
        b2[1] = Bmat[(size_t)(k2 + 2 * half + 1) * N + n];
        acc1 = __builtin_amdgcn_wmma_f32_16x16x4_f32(false, a2, false, b2,
                                                     (short)0, acc1, false, false);
    }
#pragma unroll
    for (int v = 0; v < 8; ++v) {
        const float c    = acc0[v] + acc1[v];
        const int   orow = mtile * 16 + v + 8 * half;   // C/D layout
        float val;
        if      (MODE == 0) val = fmaxf(c + bias[n], 0.f);
        else if (MODE == 1) val = c + bias[n];
        else if (MODE == 2) val = c * extra[n];
        else                val = c + extra[(size_t)orow * N + n];
        out[(size_t)orow * N + n] = val;
    }
}

// ---------------------------------------------------------------------------
// Convert x [64,1024] f32 -> bf16 (halves activation re-read traffic and
// feeds the 16-bit A fragments as two contiguous 16B chunks per lane).
// ---------------------------------------------------------------------------
__global__ void k_cvt_bf16(const float* __restrict__ x, __bf16* __restrict__ xb) {
    const int i = blockIdx.x * 256 + threadIdx.x;
    xb[i] = (__bf16)x[i];
}

// ---------------------------------------------------------------------------
// Vocab GEMM: logits[64, 50257] = x @ Ws + bs, bf16 WMMA 16x16x32, f32 acc.
// Each wave owns 2 adjacent N-tiles and all 4 M-tiles -> Ws streamed from
// HBM exactly once (non-temporal, single-use), f32->bf16 in registers.
// Boundary handling: clamp the column index instead of predicating the load
// (columns >= V are computed into never-stored accumulator lanes), so all
// loads are unconditional and batch into clauses with a single wait -- no
// per-element exec-mask/wait serialization.
// ---------------------------------------------------------------------------
__global__ void k_vocab(const float* __restrict__ Ws,
                        const float* __restrict__ bs,
                        const __bf16* __restrict__ xb,
                        float* __restrict__ out) {
    const int lane = threadIdx.x & 31;
    const int wid  = blockIdx.x * (blockDim.x >> 5) + (threadIdx.x >> 5);
    const int NPAIR = ((VOCAB + 15) / 16) / 2;   // 3142/2 = 1571
    if (wid >= NPAIR) return;                    // wave-uniform

    const int half = lane >> 4;
    const int lm   = lane & 15;
    int n[2];
    n[0] = wid * 32 + lm;
    n[1] = n[0] + 16;
    // In-bounds clamped columns for loads (duplicates are never stored).
    const int ncl0 = (n[0] < VOCAB) ? n[0] : (VOCAB - 1);
    const int ncl1 = (n[1] < VOCAB) ? n[1] : (VOCAB - 1);
    // Lane-half base: lanes 16-31 carry K+16 rows of the B fragment.
    const float* pB0 = Ws + (size_t)(16 * half) * VOCAB + ncl0;
    const float* pB1 = Ws + (size_t)(16 * half) * VOCAB + ncl1;

    v8f acc[2][4];
#pragma unroll
    for (int t = 0; t < 2; ++t)
#pragma unroll
        for (int mt = 0; mt < 4; ++mt) acc[t][mt] = (v8f){};

    for (int k = 0; k < DIM; k += 32) {
        // A fragments (16-bit A 16x32): lane = M, two contiguous 16B chunks:
        //   K = k + 16*half + [0..7]  and  K = k + 16*half + 8 + [0..7]
        v16bf af[4];
#pragma unroll
        for (int mt = 0; mt < 4; ++mt) {
            const int arow = mt * 16 + lm;
            union { uint4 q[2]; v16bf v; } u;
            u.q[0] = *(const uint4*)(xb + (size_t)arow * DIM + k + 16 * half);
            u.q[1] = *(const uint4*)(xb + (size_t)arow * DIM + k + 16 * half + 8);
            af[mt] = u.v;
        }
        // B fragments (16-bit B 32x16) for both N-tiles, unconditional NT
        // loads, then pairwise f32->bf16 packing. VGPR v packs
        // K = k + 16*half + 2v, 2v+1, lane = N.
        v16bf bf[2];
#pragma unroll
        for (int t = 0; t < 2; ++t) {
            const float* p = (t == 0 ? pB0 : pB1) + (size_t)k * VOCAB;
#pragma unroll
            for (int v = 0; v < 8; ++v) {
                const float f0 =
                    __builtin_nontemporal_load(&p[(size_t)(2 * v) * VOCAB]);
                const float f1 =
                    __builtin_nontemporal_load(&p[(size_t)(2 * v + 1) * VOCAB]);
                bf[t][2 * v]     = (__bf16)f0;
                bf[t][2 * v + 1] = (__bf16)f1;
            }
        }
#pragma unroll
        for (int t = 0; t < 2; ++t)
#pragma unroll
            for (int mt = 0; mt < 4; ++mt)
                acc[t][mt] = __builtin_amdgcn_wmma_f32_16x16x32_bf16(
                    false, af[mt], false, bf[t], (short)0, acc[t][mt], false, false);
    }
#pragma unroll
    for (int t = 0; t < 2; ++t) {
        if (n[t] < VOCAB) {
            const float bias = bs[n[t]];
#pragma unroll
            for (int mt = 0; mt < 4; ++mt)
#pragma unroll
                for (int v = 0; v < 8; ++v) {
                    const int row = mt * 16 + v + 8 * half;
                    out[(size_t)row * VOCAB + n[t]] = acc[t][mt][v] + bias;
                }
        }
    }
}

// ---------------------------------------------------------------------------
// Row softmax over V=50257, in place on d_out. One block per row.
// ---------------------------------------------------------------------------
__global__ void k_softmax(float* __restrict__ out) {
    __shared__ float red[256];
    const int tid = threadIdx.x;
    float* row = out + (size_t)blockIdx.x * VOCAB;

    float m = -INFINITY;
    for (int i = tid; i < VOCAB; i += 256) m = fmaxf(m, row[i]);
    red[tid] = m; __syncthreads();
    for (int s = 128; s > 0; s >>= 1) {
        if (tid < s) red[tid] = fmaxf(red[tid], red[tid + s]);
        __syncthreads();
    }
    m = red[0]; __syncthreads();

    float s = 0.f;
    for (int i = tid; i < VOCAB; i += 256) s += __expf(row[i] - m);
    red[tid] = s; __syncthreads();
    for (int st = 128; st > 0; st >>= 1) {
        if (tid < st) red[tid] += red[tid + st];
        __syncthreads();
    }
    const float inv = 1.f / red[0];
    for (int i = tid; i < VOCAB; i += 256) row[i] = __expf(row[i] - m) * inv;
}

// ---------------------------------------------------------------------------
// Host launcher
// ---------------------------------------------------------------------------
extern "C" void kernel_launch(void* const* d_in, const int* in_sizes, int n_in,
                              void* d_out, int out_size, void* d_ws, size_t ws_size,
                              hipStream_t stream) {
    const float*   eh  = (const float*)d_in[0];    // [64,512,1024]
    const int*     qst = (const int*)d_in[1];      // [64]
    const uint8_t* km  = (const uint8_t*)d_in[2];  // [64,512] bool
    const float*   emb = (const float*)d_in[3];    // [V,1024]
    const float*   W1  = (const float*)d_in[4];    // [1024,2048]
    const float*   b1  = (const float*)d_in[5];
    const float*   W2  = (const float*)d_in[6];    // [2048,1024]
    const float*   b2  = (const float*)d_in[7];
    const float*   Am  = (const float*)d_in[8];    // [1024,1024]
    const float*   Hm  = (const float*)d_in[9];    // [1024,1024]
    const float*   Ws  = (const float*)d_in[10];   // [1024,V]
    const float*   bs  = (const float*)d_in[11];

    float* ws = (float*)d_ws;
    float* ent_sum = ws;                        // 1024
    float* part    = ws + 1024;                 // 64*1024
    float* h1      = ws + 66560;                // 64*2048
    float* q       = ws + 197632;               // 64*1024
    float* u       = ws + 263168;               // 64*1024
    float* x       = ws + 328704;               // 64*1024
    __bf16* xb     = (__bf16*)(ws + 394240);    // 64*1024 bf16

    float* out = (float*)d_out;                 // [64, 50257]

    // 1) ent_sum = sum of masked entity hiddens (two-stage, deterministic)
    k_ent_partial<<<dim3(DIM / 256, 64), 256, 0, stream>>>(eh, km, part);
    k_ent_reduce<<<DIM / 256, 256, 0, stream>>>(part, ent_sum);

    // 2) h1 = relu(emb[question] @ W1 + b1)   : 512 tiles -> 64 blocks
    k_gemm64<0><<<64, 256, 0, stream>>>(nullptr, W1, b1, nullptr, qst, emb,
                                        h1, DIM, FC);
    // 3) q = h1 @ W2 + b2                      : 256 tiles -> 32 blocks
    k_gemm64<1><<<32, 256, 0, stream>>>(h1, W2, b2, nullptr, nullptr, nullptr,
                                        q, FC, DIM);
    // 4) u = (q @ A) * ent_sum
    k_gemm64<2><<<32, 256, 0, stream>>>(q, Am, nullptr, ent_sum, nullptr, nullptr,
                                        u, DIM, DIM);
    // 5) x = u @ H + q
    k_gemm64<3><<<32, 256, 0, stream>>>(u, Hm, nullptr, q, nullptr, nullptr,
                                        x, DIM, DIM);
    // 6) x -> bf16
    k_cvt_bf16<<<(BATCH * DIM) / 256, 256, 0, stream>>>(x, xb);

    // 7) logits = x @ Ws + bs (bf16 WMMA, Ws streamed once)
    const int npair  = ((VOCAB + 15) / 16) / 2;        // 1571 waves
    const int blocks = (npair + 7) / 8;                // 8 waves / block
    k_vocab<<<blocks, 256, 0, stream>>>(Ws, bs, xb, out);

    // 8) softmax rows, in place
    k_softmax<<<BATCH, 256, 0, stream>>>(out);
}
